// NodeGNNEncoder_6622839570791
// MI455X (gfx1250) — matrix operand order repaired and checked by reading, not compiled, not run on hardware
//
#include <hip/hip_runtime.h>
#include <hip/hip_bf16.h>

// ---------------------------------------------------------------------------
// GraphSAGE encoder for MI455X (gfx1250, wave32, WMMA).
// GEMMs run on v_wmma_f32_16x16x32_f16 with an explicit 2-deep software
// pipeline (double-buffered fragments) so loadcnt waits stagger instead of
// draining; aggregation via fp32 global atomics; bias+SiLU+LayerNorm fused
// into the GEMM epilogue through LDS.
// ---------------------------------------------------------------------------

#define N_NODES 50000
#define N_EDGES 400000
#define IN_DIM  8
#define HID     768
#define N_LAYERS 4
#define LN_EPS  1e-5f

#define KSTEP 32
#define ROWS_PER_BLK 16
#define BLOCK_T 512
#define WAVES_PER_BLK 16
#define TILES_PER_WAVE 3           // 768/16 = 48 N-tiles / 16 waves
#define SMEM_STRIDE 770            // 768 + 2: breaks 64-bank aliasing between row j and j+8

typedef __attribute__((ext_vector_type(16))) _Float16 v16h;
typedef __attribute__((ext_vector_type(8)))  _Float16 v8h;
typedef __attribute__((ext_vector_type(4)))  _Float16 v4h;
typedef __attribute__((ext_vector_type(8)))  float    v8f;

// ------------------------------ utility kernels -----------------------------

__global__ void zero_f32_kernel(float* __restrict__ p, long long n) {
    long long i = (long long)blockIdx.x * blockDim.x + threadIdx.x;
    if (i < n) p[i] = 0.0f;
}

__global__ void degree_kernel(const int* __restrict__ dst, float* __restrict__ deg, int e) {
    int i = blockIdx.x * blockDim.x + threadIdx.x;
    if (i < e) atomicAdd(&deg[dst[i]], 1.0f);
}

__global__ void invdeg_kernel(float* __restrict__ deg, int n) {
    int i = blockIdx.x * blockDim.x + threadIdx.x;
    if (i < n) deg[i] = 1.0f / fmaxf(deg[i], 1.0f);
}

// fp32 -> f16 (weights; done once, amortized over 3125 blocks x 4 layers)
__global__ void f32_to_f16_kernel(const float* __restrict__ w, _Float16* __restrict__ wh, long long n) {
    long long i = (long long)blockIdx.x * blockDim.x + threadIdx.x;
    if (i < n) wh[i] = (_Float16)w[i];
}

// h0 = x @ Wp.T + bp  (K = 8, memory-bound; plain VALU), stored as f16
__global__ void input_proj_kernel(const float* __restrict__ x, const float* __restrict__ Wp,
                                  const float* __restrict__ bp, _Float16* __restrict__ h) {
    int i = blockIdx.x * blockDim.x + threadIdx.x;   // over N*HID
    if (i >= N_NODES * HID) return;
    int n = i / HID, c = i - n * HID;
    const float* xr = x + (size_t)n * IN_DIM;
    const float* wr = Wp + (size_t)c * IN_DIM;
    float s = bp[c];
#pragma unroll
    for (int k = 0; k < IN_DIM; ++k) s += xr[k] * wr[k];
    h[i] = (_Float16)s;
}

// agg[dst] += h[src]   (4 channels per thread, fp32 atomics)
__global__ void scatter_kernel(const _Float16* __restrict__ h, const int* __restrict__ src,
                               const int* __restrict__ dst, float* __restrict__ agg) {
    int i = blockIdx.x * blockDim.x + threadIdx.x;   // E * (HID/4) threads
    int e = i / (HID / 4);
    int c = (i - e * (HID / 4)) * 4;
    int s = src[e], d = dst[e];
    v4h hv = *(const v4h*)(h + (size_t)s * HID + c);
    float* ap = agg + (size_t)d * HID + c;
#pragma unroll
    for (int j = 0; j < 4; ++j) atomicAdd(ap + j, (float)hv[j]);
}

// agg_h = (f16)(agg * inv_deg[row])
__global__ void agg_scale_kernel(const float* __restrict__ agg, const float* __restrict__ invd,
                                 _Float16* __restrict__ aggh) {
    int i = blockIdx.x * blockDim.x + threadIdx.x;   // over N*HID
    if (i >= N_NODES * HID) return;
    int n = i / HID;
    aggh[i] = (_Float16)(agg[i] * invd[n]);
}

// --------------------------- fused GEMM + SiLU + LN -------------------------
// out = silu(aggh @ Wl.T + bl + h @ Wr.T) -> LayerNorm(gamma, beta)
// Block: 512 threads (16 waves), computes 16 rows x 768 cols.

// Load A fragment (contiguous K order, lane half h: k0+16h..k0+16h+15) for K-step k0.
#define LOAD_A(dsth, dstg, k0)                                                        \
    do {                                                                              \
        v8h _hl = *(const v8h*)(ahp + (k0));                                          \
        v8h _hh = *(const v8h*)(ahp + (k0) + 8);                                      \
        v8h _gl = *(const v8h*)(agp + (k0));                                          \
        v8h _gh = *(const v8h*)(agp + (k0) + 8);                                      \
        dsth = __builtin_shufflevector(_hl, _hh, 0,1,2,3,4,5,6,7,8,9,10,11,12,13,14,15); \
        dstg = __builtin_shufflevector(_gl, _gh, 0,1,2,3,4,5,6,7,8,9,10,11,12,13,14,15); \
    } while (0)

// Load B fragments (lane<16: K=k0..k0+15; lane>=16: K=k0+16..k0+31 via bkb offset).
#define LOAD_B(dl, dr, k0)                                                            \
    do {                                                                              \
        _Pragma("unroll")                                                             \
        for (int _t = 0; _t < TILES_PER_WAVE; ++_t) {                                 \
            dl[_t] = *(const v16h*)(blp[_t] + (k0));                                  \
            dr[_t] = *(const v16h*)(brp[_t] + (k0));                                  \
        }                                                                             \
    } while (0)

#define DO_WMMA(ag, ah, dl, dr)                                                       \
    do {                                                                              \
        _Pragma("unroll")                                                             \
        for (int _t = 0; _t < TILES_PER_WAVE; ++_t) {                                 \
            acc[_t] = __builtin_amdgcn_wmma_f32_16x16x32_f16(                         \
                false, ag, false, dl[_t], (short)0, acc[_t], false, false);           \
            acc[_t] = __builtin_amdgcn_wmma_f32_16x16x32_f16(                         \
                false, ah, false, dr[_t], (short)0, acc[_t], false, false);           \
        }                                                                             \
    } while (0)

__global__ __launch_bounds__(BLOCK_T, 1)
void gemm_ln_kernel(const _Float16* __restrict__ Ah,   // hidden state (root term)
                    const _Float16* __restrict__ Ag,   // scaled aggregation
                    const _Float16* __restrict__ Bl,   // Wl (f16, row-major [n][k])
                    const _Float16* __restrict__ Br,   // Wr (f16, row-major [n][k])
                    const float* __restrict__ bl,
                    const float* __restrict__ gamma, const float* __restrict__ beta,
                    _Float16* __restrict__ hout, float* __restrict__ fout) {
    __shared__ float smem[ROWS_PER_BLK * SMEM_STRIDE];
    __shared__ float s_mu[ROWS_PER_BLK];
    __shared__ float s_rs[ROWS_PER_BLK];

    const int lane = threadIdx.x & 31;
    const int wave = threadIdx.x >> 5;     // 0..15
    const int lrow = lane & 15;
    const int hi   = lane >> 4;            // 0 / 1 half-wave
    const int akb  = hi * 16;              // A frag: K sub-offset (contiguous K order, matches B)
    const int bkb  = hi * 16;              // B frag: K sub-offset (contiguous K order, matches A)
    const int m0   = blockIdx.x * ROWS_PER_BLK;

    v8f acc[TILES_PER_WAVE] = {};

    // Per-tile weight row pointers (n depends only on tile, not k)
    const _Float16* blp[TILES_PER_WAVE];
    const _Float16* brp[TILES_PER_WAVE];
#pragma unroll
    for (int t = 0; t < TILES_PER_WAVE; ++t) {
        const int n = (wave * TILES_PER_WAVE + t) * 16 + lrow;
        blp[t] = Bl + (size_t)n * HID + bkb;
        brp[t] = Br + (size_t)n * HID + bkb;
    }
    const _Float16* ahp = Ah + (size_t)(m0 + lrow) * HID + akb;
    const _Float16* agp = Ag + (size_t)(m0 + lrow) * HID + akb;

    // -------- explicit 2-deep software pipeline over K (two steps / iter) ----
    v16h a0h, a0g, a1h, a1g;
    v16h bl0[TILES_PER_WAVE], br0[TILES_PER_WAVE];
    v16h bl1[TILES_PER_WAVE], br1[TILES_PER_WAVE];

    LOAD_B(bl0, br0, 0);
    LOAD_A(a0h, a0g, 0);
#pragma unroll 1
    for (int k0 = 0; k0 < HID; k0 += 2 * KSTEP) {
        // issue loads for k0+32 while computing k0 (k0+32 <= 736 always in range)
        LOAD_B(bl1, br1, k0 + KSTEP);
        LOAD_A(a1h, a1g, k0 + KSTEP);
        __builtin_prefetch(ahp + k0 + 2 * KSTEP, 0, 3);   // near-scope prefetch of
        __builtin_prefetch(agp + k0 + 2 * KSTEP, 0, 3);   // HBM-streaming A operand
        DO_WMMA(a0g, a0h, bl0, br0);

        if (k0 + 2 * KSTEP < HID) {                       // uniform branch, EXEC intact
            LOAD_B(bl0, br0, k0 + 2 * KSTEP);
            LOAD_A(a0h, a0g, k0 + 2 * KSTEP);
        }
        DO_WMMA(a1g, a1h, bl1, br1);
    }

    // bias + SiLU -> LDS tile (C layout: VGPR j -> row j + 8*hi, col = tile*16 + lrow)
#pragma unroll
    for (int t = 0; t < TILES_PER_WAVE; ++t) {
        const int n = (wave * TILES_PER_WAVE + t) * 16 + lrow;
        const float blv = bl[n];
#pragma unroll
        for (int j = 0; j < 8; ++j) {
            float v = acc[t][j] + blv;
            float s = v / (1.0f + __expf(-v));      // silu
            smem[(j + hi * 8) * SMEM_STRIDE + n] = s;
        }
    }
    __syncthreads();

    // per-row mean / rstd: each wave reduces its row of 768 with shfl butterflies
    {
        const int r = wave;                          // 16 waves, 16 rows
        float s = 0.0f, s2 = 0.0f;
        for (int c = lane; c < HID; c += 32) {
            float v = smem[r * SMEM_STRIDE + c];
            s += v; s2 += v * v;
        }
#pragma unroll
        for (int off = 16; off > 0; off >>= 1) {
            s  += __shfl_xor(s,  off, 32);
            s2 += __shfl_xor(s2, off, 32);
        }
        if (lane == 0) {
            const float mu  = s  * (1.0f / HID);
            const float var = s2 * (1.0f / HID) - mu * mu;
            s_mu[r] = mu;
            s_rs[r] = rsqrtf(var + LN_EPS);
        }
    }
    __syncthreads();

    // normalize + affine, coalesced write of h (f16) and optionally fp32 output
    for (int idx = threadIdx.x; idx < ROWS_PER_BLK * HID; idx += BLOCK_T) {
        const int m = idx / HID, c = idx - m * HID;
        const float v = (smem[m * SMEM_STRIDE + c] - s_mu[m]) * s_rs[m] * gamma[c] + beta[c];
        const size_t o = (size_t)(m0 + m) * HID + c;
        hout[o] = (_Float16)v;
        if (fout) fout[o] = v;
    }
}

// ------------------------------- host driver --------------------------------

extern "C" void kernel_launch(void* const* d_in, const int* in_sizes, int n_in,
                              void* d_out, int out_size, void* d_ws, size_t ws_size,
                              hipStream_t stream) {
    const float* x     = (const float*)d_in[0];
    const float* Wp    = (const float*)d_in[1];
    const float* bp    = (const float*)d_in[2];
    const float* Wl    = (const float*)d_in[3];
    const float* bl    = (const float*)d_in[4];
    const float* Wr    = (const float*)d_in[5];
    const float* gamma = (const float*)d_in[6];
    const float* beta  = (const float*)d_in[7];
    const int*   ei    = (const int*)d_in[8];
    const int* src = ei;
    const int* dst = ei + N_EDGES;
    float* out = (float*)d_out;

    // workspace carve-up
    char* ws = (char*)d_ws;
    size_t off = 0;
    auto carve = [&](size_t bytes) { size_t p = off; off += (bytes + 255) & ~(size_t)255; return p; };
    const size_t NH = (size_t)N_NODES * HID;
    float*    deg   = (float*)   (ws + carve((size_t)N_NODES * 4));
    float*    agg   = (float*)   (ws + carve(NH * 4));
    _Float16* aggh  = (_Float16*)(ws + carve(NH * 2));
    _Float16* hbuf0 = (_Float16*)(ws + carve(NH * 2));
    _Float16* hbuf1 = (_Float16*)(ws + carve(NH * 2));
    const size_t WN = (size_t)N_LAYERS * HID * HID;
    _Float16* wlh   = (_Float16*)(ws + carve(WN * 2));
    _Float16* wrh   = (_Float16*)(ws + carve(WN * 2));
    _Float16* hping[2] = {hbuf0, hbuf1};

    const int T = 256;
    auto blks = [](long long n, int t) { return (int)((n + t - 1) / t); };

    // degrees -> inverse degrees
    zero_f32_kernel<<<blks(N_NODES, T), T, 0, stream>>>(deg, N_NODES);
    degree_kernel<<<blks(N_EDGES, T), T, 0, stream>>>(dst, deg, N_EDGES);
    invdeg_kernel<<<blks(N_NODES, T), T, 0, stream>>>(deg, N_NODES);

    // weights -> f16
    f32_to_f16_kernel<<<blks(WN, T), T, 0, stream>>>(Wl, wlh, (long long)WN);
    f32_to_f16_kernel<<<blks(WN, T), T, 0, stream>>>(Wr, wrh, (long long)WN);

    // input projection
    input_proj_kernel<<<blks(NH, T), T, 0, stream>>>(x, Wp, bp, hping[0]);

    const long long scatter_threads = (long long)N_EDGES * (HID / 4);
    for (int l = 0; l < N_LAYERS; ++l) {
        _Float16* hin  = hping[l & 1];
        _Float16* hout = hping[(l + 1) & 1];
        zero_f32_kernel<<<blks(NH, T), T, 0, stream>>>(agg, (long long)NH);
        scatter_kernel<<<blks(scatter_threads, T), T, 0, stream>>>(hin, src, dst, agg);
        agg_scale_kernel<<<blks(NH, T), T, 0, stream>>>(agg, deg, aggh);
        gemm_ln_kernel<<<N_NODES / ROWS_PER_BLK, BLOCK_T, 0, stream>>>(
            hin, aggh,
            wlh + (size_t)l * HID * HID, wrh + (size_t)l * HID * HID,
            bl + (size_t)l * HID, gamma + (size_t)l * HID, beta + (size_t)l * HID,
            hout, (l == N_LAYERS - 1) ? out : nullptr);
    }
}
